// MultiHeadAttention_82772609729207
// MI455X (gfx1250) — matrix operand order repaired and checked by reading, not compiled, not run on hardware
//
#include <hip/hip_runtime.h>

// ---------------------------------------------------------------------------
// MI455X / gfx1250 multi-head attention: bf16 WMMA GEMMs + flash attention.
// Round 4: GEMM now stages A/B k-slices in LDS via the CDNA5 async path
// (global_load_async_to_lds_b128 + s_wait_asynccnt), double-buffered, with
// fragments served from LDS (ds_load_b128). Cuts per-block VMEM traffic 3x
// and overlaps global latency with the XDL pipe.
// ---------------------------------------------------------------------------

typedef __attribute__((ext_vector_type(16))) __bf16 v16bf;
typedef __attribute__((ext_vector_type(8)))  float  v8f;

#define B_DIM 2
#define S_DIM 2048
#define D_DIM 1024
#define H_DIM 16
#define HD_DIM 64
#define M_DIM (B_DIM * S_DIM)   // 4096

// ---- helpers ---------------------------------------------------------------

__device__ __forceinline__ unsigned short f2bf_bits(float f) {
    unsigned u = __float_as_uint(f);
    unsigned r = u + 0x7FFFu + ((u >> 16) & 1u);   // round-to-nearest-even
    return (unsigned short)(r >> 16);
}
__device__ __forceinline__ __bf16 f2bf(float f) {
    unsigned short b = f2bf_bits(f);
    __bf16 r;
    __builtin_memcpy(&r, &b, 2);
    return r;
}

// A-fragment (16x32, 16-bit): lane holds row M = lane&15;
// lane-half kh = lane>>4 holds K = kh*16 .. kh*16+15 (contiguous, matches B-frag).
__device__ __forceinline__ v16bf load_a_frag(const __bf16* rowp, int kh) {
    union { v16bf v; uint4 q[2]; } u;
    u.q[0] = *reinterpret_cast<const uint4*>(rowp + kh * 16);
    u.q[1] = *reinterpret_cast<const uint4*>(rowp + kh * 16 + 8);
    return u.v;
}

// B-fragment (32x16, 16-bit): lane holds column N = lane&15; lanes 0-15 hold
// K=0..15, lanes 16-31 hold K=16..31 (contiguous 32B from B^T row-major).
__device__ __forceinline__ v16bf load_b_frag(const __bf16* p) {
    union { v16bf v; uint4 q[2]; } u;
    u.q[0] = reinterpret_cast<const uint4*>(p)[0];
    u.q[1] = reinterpret_cast<const uint4*>(p)[1];
    return u.v;
}

__device__ __forceinline__ v8f wmma_bf16(v16bf a, v16bf b, v8f c) {
    return __builtin_amdgcn_wmma_f32_16x16x32_bf16(
        /*neg_a=*/false, a, /*neg_b=*/false, b,
        /*c_mod=*/(short)0, c, /*reuse_a=*/false, /*reuse_b=*/false);
}

// Async copy of one 16B chunk from global to LDS (CDNA5 Tensor/async path,
// tracked by ASYNCcnt; LDS address = low 32 bits of the generic address,
// as the ISA's LDS aperture rule gives).
__device__ __forceinline__ void async_g2l_b128(void* lds_ptr, const void* gptr) {
    unsigned la = (unsigned)(unsigned long long)lds_ptr;
    asm volatile("global_load_async_to_lds_b128 %0, %1, off"
                 :: "v"(la), "v"(gptr) : "memory");
}
__device__ __forceinline__ void wait_async0() {
    asm volatile("s_wait_asynccnt 0x0" ::: "memory");
}

// ---- elementwise f32 -> bf16 ----------------------------------------------

__global__ void cvt_bf16_kernel(const float* __restrict__ in,
                                __bf16* __restrict__ out, int n) {
    int i = blockIdx.x * blockDim.x + threadIdx.x;
    if (i < n) out[i] = f2bf(in[i]);
}

// ---- f32 [rows,cols] -> bf16 transposed [cols,rows] ------------------------

__global__ __launch_bounds__(256)
void transpose_cvt_kernel(const float* __restrict__ in,
                          __bf16* __restrict__ out, int rows, int cols) {
    __shared__ float tile[32][33];
    int x = blockIdx.x * 32 + threadIdx.x;   // input col
    int y = blockIdx.y * 32 + threadIdx.y;   // input row
    #pragma unroll
    for (int j = 0; j < 32; j += 8)
        tile[threadIdx.y + j][threadIdx.x] = in[(size_t)(y + j) * cols + x];
    __syncthreads();
    int ox = blockIdx.y * 32 + threadIdx.x;  // output col = input row
    int oy = blockIdx.x * 32 + threadIdx.y;  // output row = input col
    #pragma unroll
    for (int j = 0; j < 32; j += 8)
        out[(size_t)(oy + j) * rows + ox] = f2bf(tile[threadIdx.x][threadIdx.y + j]);
}

// ---- WMMA GEMM: C[M,N] = A[M,K] (bf16 row-major) x B (given as B^T [N,K]) --
// Block: 256 threads = 8 waves (2 M x 4 N), block tile 128x128, wave 64x32.
// K-slices of 32 staged in LDS via async loads, double-buffered.
// MODE 0: bf16 out, head-split   [B,H,S,hd]
// MODE 2: bf16 out, head-split-T [B,H,hd,S]
// MODE 3: f32 out row-major [M,N] + bias[n]

#define LDT 40   // LDS row stride in bf16 (80B: 16B-aligned, bank-skewed)

template <int MODE>
__global__ __launch_bounds__(256)
void gemm_bf16_kernel(const __bf16* __restrict__ A, const __bf16* __restrict__ BT,
                      void* __restrict__ Cout, const float* __restrict__ bias,
                      int Mdim, int Ndim, int Kdim) {
    __shared__ alignas(16) __bf16 sA[2][128 * LDT];
    __shared__ alignas(16) __bf16 sB[2][128 * LDT];

    const int tid  = threadIdx.x;
    const int lane = tid & 31;
    const int wave = tid >> 5;
    const int wm = wave & 1;
    const int wn = wave >> 1;
    const int l15 = lane & 15;
    const int kh  = lane >> 4;
    const int m0 = blockIdx.y * 128;
    const int n0 = blockIdx.x * 128;

    // Cooperative async staging of one 128x32 bf16 slice of A and of B^T:
    // 512 chunks of 16B each; each thread issues 2 chunks per matrix.
    auto prefetch = [&](int k, int buf) {
        for (int c = tid; c < 512; c += 256) {
            int row = c >> 2;
            int ch  = (c & 3) * 8;                      // bf16 column of chunk
            async_g2l_b128(&sA[buf][row * LDT + ch],
                           A  + (size_t)(m0 + row) * Kdim + k + ch);
            async_g2l_b128(&sB[buf][row * LDT + ch],
                           BT + (size_t)(n0 + row) * Kdim + k + ch);
        }
    };

    v8f acc[4][2];
    #pragma unroll
    for (int i = 0; i < 4; ++i)
        #pragma unroll
        for (int j = 0; j < 2; ++j) acc[i][j] = 0.0f;

    prefetch(0, 0);
    wait_async0();
    __syncthreads();

    const int nk = Kdim / 32;
    for (int ki = 0; ki < nk; ++ki) {
        const int buf = ki & 1;
        if (ki + 1 < nk) prefetch((ki + 1) * 32, buf ^ 1);

        const __bf16* pa = sA[buf];
        const __bf16* pb = sB[buf];
        v16bf af[4], bfr[2];
        #pragma unroll
        for (int i = 0; i < 4; ++i)
            af[i] = load_a_frag(pa + (wm * 64 + i * 16 + l15) * LDT, kh);
        #pragma unroll
        for (int j = 0; j < 2; ++j)
            bfr[j] = load_b_frag(pb + (wn * 32 + j * 16 + l15) * LDT + kh * 16);
        #pragma unroll
        for (int i = 0; i < 4; ++i)
            #pragma unroll
            for (int j = 0; j < 2; ++j)
                acc[i][j] = wmma_bf16(af[i], bfr[j], acc[i][j]);

        // next slice resident + all waves done reading this buffer
        wait_async0();
        __syncthreads();
    }

    // Epilogue: single compile-time mode, store each tile immediately.
    #pragma unroll
    for (int i = 0; i < 4; ++i)
        #pragma unroll
        for (int j = 0; j < 2; ++j) {
            const int mbase = m0 + wm * 64 + i * 16 + 8 * kh;   // + r
            const int n     = n0 + wn * 32 + j * 16 + l15;      // C/D col
            #pragma unroll
            for (int r = 0; r < 8; ++r) {
                int m = mbase + r;
                float v = acc[i][j][r];
                if (MODE == 3) {
                    ((float*)Cout)[(size_t)m * Ndim + n] = v + bias[n];
                } else {
                    int b = m >> 11, s = m & (S_DIM - 1);
                    int h = n >> 6,  dh = n & (HD_DIM - 1);
                    size_t idx;
                    if (MODE == 0)
                        idx = ((size_t)((b * H_DIM + h) * S_DIM + s) << 6) + dh;
                    else  // MODE 2: transposed V
                        idx = (size_t)((b * H_DIM + h) * HD_DIM + dh) * S_DIM + s;
                    ((__bf16*)Cout)[idx] = f2bf(v);
                }
            }
        }
}

// ---- Flash attention (causal) ----------------------------------------------
// Q,K: bf16 [B,H,S,hd]; Vt: bf16 [B,H,hd,S]; ctx out: bf16 [B,S,D].
// 1 wave handles 16 q-rows; kv tiles of 32; online softmax.
// MASKED=false: tile fully visible (kv+31 <= q0), no causal compares.

template <bool MASKED>
__device__ __forceinline__ void flash_step(
    int kv, int q0, int col, int kh,
    const __bf16* __restrict__ Kbh, const __bf16* __restrict__ Vbh,
    v16bf aq0, v16bf aq1, v16bf ones, unsigned short* pb,
    float (&mrun)[8], float (&lrun)[8], v8f (&cacc)[4]) {
    const float L2E = 1.44269504f;

    // ---- scores: two 16x16 tiles over kv, accumulated over hd=64 ----
    v8f s0 = 0.0f, s1 = 0.0f;
    {
        const __bf16* k0p = Kbh + (size_t)(kv + col) * HD_DIM + kh * 16;
        const __bf16* k1p = Kbh + (size_t)(kv + 16 + col) * HD_DIM + kh * 16;
        v16bf b00 = load_b_frag(k0p);        // kv tile0, hd 0..31
        v16bf b01 = load_b_frag(k0p + 32);   // kv tile0, hd 32..63
        v16bf b10 = load_b_frag(k1p);
        v16bf b11 = load_b_frag(k1p + 32);
        s0 = wmma_bf16(aq0, b00, s0);
        s0 = wmma_bf16(aq1, b01, s0);
        s1 = wmma_bf16(aq0, b10, s1);
        s1 = wmma_bf16(aq1, b11, s1);
    }

    // ---- online softmax: row max (shuffle reduce), exp, store P ----
    float corr[8];
    #pragma unroll
    for (int r = 0; r < 8; ++r) {
        float a = s0[r] * 0.125f;                        // 1/sqrt(64)
        float c = s1[r] * 0.125f;
        if (MASKED) {
            int qrow_g = q0 + r + 8 * kh;
            if (kv + col      > qrow_g) a = -1e30f;      // causal mask
            if (kv + 16 + col > qrow_g) c = -1e30f;
        }
        float mx = fmaxf(a, c);
        mx = fmaxf(mx, __shfl_xor(mx, 1));
        mx = fmaxf(mx, __shfl_xor(mx, 2));
        mx = fmaxf(mx, __shfl_xor(mx, 4));
        mx = fmaxf(mx, __shfl_xor(mx, 8));
        float mnew = fmaxf(mrun[r], mx);
        float pa = __builtin_amdgcn_exp2f((a - mnew) * L2E);
        float pc = __builtin_amdgcn_exp2f((c - mnew) * L2E);
        corr[r] = __builtin_amdgcn_exp2f((mrun[r] - mnew) * L2E);
        mrun[r] = mnew;
        int row = r + 8 * kh;
        pb[row * 32 + col]      = f2bf_bits(pa);
        pb[row * 32 + 16 + col] = f2bf_bits(pc);
    }
    #pragma unroll
    for (int t = 0; t < 4; ++t)
        #pragma unroll
        for (int r = 0; r < 8; ++r) cacc[t][r] *= corr[r];

    // LDS stores -> loads within the same wave; keep order explicit
    __builtin_amdgcn_wave_barrier();
    asm volatile("s_wait_dscnt 0" ::: "memory");

    // ---- re-load P in A-fragment layout (16 q x 32 kv) ----
    v16bf pf;
    {
        const unsigned short* rp = pb + col * 32;
        union { v16bf v; uint4 q[2]; } u;
        u.q[0] = *reinterpret_cast<const uint4*>(rp + kh * 16);
        u.q[1] = *reinterpret_cast<const uint4*>(rp + kh * 16 + 8);
        pf = u.v;
    }

    // ---- row sums via WMMA against ones: D[m][n] = rowsum(P[m]) ----
    v8f sums = wmma_bf16(pf, ones, (v8f)0.0f);
    #pragma unroll
    for (int r = 0; r < 8; ++r)
        lrun[r] = lrun[r] * corr[r] + sums[r];

    // ---- ctx += P @ V : 4 hd tiles of 16 ----
    #pragma unroll
    for (int t = 0; t < 4; ++t) {
        const __bf16* vr = Vbh + (size_t)(t * 16 + col) * S_DIM + kv + kh * 16;
        v16bf bv = load_b_frag(vr);
        cacc[t] = wmma_bf16(pf, bv, cacc[t]);
    }
    __builtin_amdgcn_wave_barrier();
}

__global__ __launch_bounds__(128)
void flash_attn_kernel(const __bf16* __restrict__ Q, const __bf16* __restrict__ K,
                       const __bf16* __restrict__ Vt, __bf16* __restrict__ ctx) {
    const int lane = threadIdx.x & 31;
    const int wave = threadIdx.x >> 5;
    const int col  = lane & 15;
    const int kh   = lane >> 4;
    const int b = blockIdx.z, h = blockIdx.y;
    const int q0 = blockIdx.x * 64 + wave * 16;

    const __bf16* Qbh = Q  + (size_t)(b * H_DIM + h) * S_DIM * HD_DIM;
    const __bf16* Kbh = K  + (size_t)(b * H_DIM + h) * S_DIM * HD_DIM;
    const __bf16* Vbh = Vt + (size_t)(b * H_DIM + h) * HD_DIM * S_DIM;

    // Q A-fragments for hd 0..31 and 32..63 (loaded once)
    const __bf16* qrow = Qbh + (size_t)(q0 + col) * HD_DIM;
    v16bf aq0 = load_a_frag(qrow, kh);
    v16bf aq1 = load_a_frag(qrow + 32, kh);

    // ones fragment (bf16 1.0 = 0x3F80)
    v16bf ones;
    {
        union { v16bf v; unsigned q[8]; } u;
        #pragma unroll
        for (int i = 0; i < 8; ++i) u.q[i] = 0x3F803F80u;
        ones = u.v;
    }

    float mrun[8], lrun[8];
    v8f cacc[4];
    #pragma unroll
    for (int r = 0; r < 8; ++r) { mrun[r] = -1e30f; lrun[r] = 0.0f; }
    #pragma unroll
    for (int t = 0; t < 4; ++t) cacc[t] = 0.0f;

    __shared__ alignas(16) unsigned short pbuf[4][16 * 32];
    unsigned short* pb = pbuf[wave];

    int kv = 0;
    // interior tiles: fully visible, no causal compares
    for (; kv + 31 <= q0; kv += 32)
        flash_step<false>(kv, q0, col, kh, Kbh, Vbh, aq0, aq1, ones, pb,
                          mrun, lrun, cacc);
    // diagonal tile(s): causal mask active
    for (; kv < q0 + 16; kv += 32)
        flash_step<true>(kv, q0, col, kh, Kbh, Vbh, aq0, aq1, ones, pb,
                         mrun, lrun, cacc);

    // ---- normalize and write ctx bf16 row-major [B,S,D] ----
    #pragma unroll
    for (int t = 0; t < 4; ++t)
        #pragma unroll
        for (int r = 0; r < 8; ++r) {
            int srow = q0 + r + 8 * kh;
            int dcol = h * HD_DIM + t * 16 + col;
            float v = cacc[t][r] / lrun[r];
            ctx[(size_t)(b * S_DIM + srow) * D_DIM + dcol] = f2bf(v);
        }
}

// ---------------------------------------------------------------------------

extern "C" void kernel_launch(void* const* d_in, const int* in_sizes, int n_in,
                              void* d_out, int out_size, void* d_ws, size_t ws_size,
                              hipStream_t stream) {
    const float* X  = (const float*)d_in[0];
    const float* Wq = (const float*)d_in[1];
    const float* Wk = (const float*)d_in[2];
    const float* Wv = (const float*)d_in[3];
    const float* Wo = (const float*)d_in[4];
    const float* bo = (const float*)d_in[5];
    float* out = (float*)d_out;

    char* ws = (char*)d_ws;
    const size_t MB = 1u << 20;
    __bf16* Xb  = (__bf16*)(ws + 0 * MB);    // 8 MB  : X bf16 [4096,1024]
    __bf16* WqT = (__bf16*)(ws + 8 * MB);    // 2 MB  : Wq^T bf16 [1024,1024]
    __bf16* WkT = (__bf16*)(ws + 10 * MB);
    __bf16* WvT = (__bf16*)(ws + 12 * MB);
    __bf16* WoT = (__bf16*)(ws + 14 * MB);
    __bf16* Qh  = (__bf16*)(ws + 16 * MB);   // 8 MB  : [B,H,S,hd]
    __bf16* Kh  = (__bf16*)(ws + 24 * MB);   // 8 MB  : [B,H,S,hd]
    __bf16* Vt  = (__bf16*)(ws + 32 * MB);   // 8 MB  : [B,H,hd,S]
    __bf16* Cb  = (__bf16*)(ws + 40 * MB);   // 8 MB  : ctx bf16 [4096,1024]

    // 1) convert X to bf16
    {
        int n = M_DIM * D_DIM;
        cvt_bf16_kernel<<<(n + 255) / 256, 256, 0, stream>>>(X, Xb, n);
    }
    // 2) transpose + convert weights
    {
        dim3 blk(32, 8), grd(D_DIM / 32, D_DIM / 32);
        transpose_cvt_kernel<<<grd, blk, 0, stream>>>(Wq, WqT, D_DIM, D_DIM);
        transpose_cvt_kernel<<<grd, blk, 0, stream>>>(Wk, WkT, D_DIM, D_DIM);
        transpose_cvt_kernel<<<grd, blk, 0, stream>>>(Wv, WvT, D_DIM, D_DIM);
        transpose_cvt_kernel<<<grd, blk, 0, stream>>>(Wo, WoT, D_DIM, D_DIM);
    }
    // 3) Q/K/V projections (head-split outputs; V transposed)
    {
        dim3 grd(D_DIM / 128, M_DIM / 128);  // (8, 32)
        gemm_bf16_kernel<0><<<grd, 256, 0, stream>>>(Xb, WqT, Qh, nullptr,
                                                     M_DIM, D_DIM, D_DIM);
        gemm_bf16_kernel<0><<<grd, 256, 0, stream>>>(Xb, WkT, Kh, nullptr,
                                                     M_DIM, D_DIM, D_DIM);
        gemm_bf16_kernel<2><<<grd, 256, 0, stream>>>(Xb, WvT, Vt, nullptr,
                                                     M_DIM, D_DIM, D_DIM);
    }
    // 4) flash attention
    {
        dim3 grd(S_DIM / 64, H_DIM, B_DIM);  // (32, 16, 2)
        flash_attn_kernel<<<grd, 128, 0, stream>>>(Qh, Kh, Vt, Cb);
    }
    // 5) output projection + bias -> f32
    {
        dim3 grd(D_DIM / 128, M_DIM / 128);
        gemm_bf16_kernel<3><<<grd, 256, 0, stream>>>(Cb, WoT, out, bo,
                                                     M_DIM, D_DIM, D_DIM);
    }
}
